// MeshLexVQVAE_14946486190090
// MI455X (gfx1250) — matrix operand order repaired and checked by reading, not compiled, not run on hardware
//
#include <hip/hip_runtime.h>
#include <math.h>

#define N_NODES 131072
#define N_EDGES 393216
#define BB      256
#define EMB     128
#define KCODE   4096
#define MAXV    128
#define LN_EPS  1e-5f
#define RECON_N (BB * MAXV * 3)

typedef __attribute__((ext_vector_type(16))) __bf16        v16bf;
typedef __attribute__((ext_vector_type(8)))  float         v8f;
typedef __attribute__((ext_vector_type(8)))  unsigned int  v8u;

__device__ __forceinline__ float gelu_f(float x) {
  return 0.5f * x * (1.0f + erff(x * 0.70710678118654752440f));
}

__device__ __forceinline__ unsigned pack_bf16(float a, float b) {
  union { __bf16 h[2]; unsigned u; } p;
  p.h[0] = (__bf16)a;
  p.h[1] = (__bf16)b;
  return p.u;   // lowers to v_cvt_pk_bf16_f32
}

union FragCast { v8u u; v16bf h; };

// ---------------------------------------------------------------------------
// Tiled WMMA GEMM:
//   C[M,N] = A[M,K] @ W[K, wcol0:wcol0+N]  (+ A2 @ W2 if DUAL) (+bias) (gelu?)
// Block tile 64x128, 8 waves (2x4), each wave owns 2x2 16x16 WMMA tiles.
// K templated (15/128/256); all epilogue options are compile-time.
// LDS holds packed-bf16 K-pairs:
//   A:  AsU[row 0..63][kpair 0..15]   (k, k+1 packed in one dword)
//   B:  BsU[col 0..127][kpair 0..15]  (transposed at staging time)
// so every fragment is exactly two ds_load_b128.
// ---------------------------------------------------------------------------
template <int KV, bool DUAL, bool HASBIAS, bool GELU>
__global__ __launch_bounds__(256) void gemm_bf16_wmma(
    const float* __restrict__ A,  const float* __restrict__ W,
    const float* __restrict__ A2, const float* __restrict__ W2,
    const float* __restrict__ bias, float* __restrict__ Cmat,
    int M, int N, int ldw, int wcol0)
{
  __shared__ unsigned AsU[64][20];    // stride 20 dwords: 16B-aligned + bank spread
  __shared__ unsigned BsU[128][20];

  const int tid  = threadIdx.x;
  const int lane = tid & 31;
  const int wave = tid >> 5;
  const int wm   = wave >> 2;         // 0..1
  const int wn   = wave & 3;          // 0..3
  const int blockM = blockIdx.y * 64;
  const int blockN = blockIdx.x * 128;

  v8f acc[2][2] = {};

  constexpr int kTiles = (KV + 31) / 32;
  constexpr int phases = DUAL ? 2 : 1;

#pragma unroll 1
  for (int ph = 0; ph < phases; ++ph) {
    const float* Ap = (DUAL && ph) ? A2 : A;
    const float* Wp = (DUAL && ph) ? W2 : W;

    for (int kt = 0; kt < kTiles; ++kt) {
      const int k0 = kt * 32;

      // ---- stage A tile (64 rows x 32 K), 512 float4 chunks, 2/thread ----
      for (int t = tid; t < 512; t += 256) {
        const int r  = t >> 3;          // 0..63
        const int ck = (t & 7) * 4;     // 0,4,...,28
        const int gk = k0 + ck;
        const float* ap = Ap + (size_t)(blockM + r) * KV + gk;
        float e0, e1, e2, e3;
        if constexpr (KV % 32 == 0) {
          float4 q = *(const float4*)ap;            // global_load_b128
          e0 = q.x; e1 = q.y; e2 = q.z; e3 = q.w;
        } else {
          e0 = (gk + 0 < KV) ? ap[0] : 0.0f;
          e1 = (gk + 1 < KV) ? ap[1] : 0.0f;
          e2 = (gk + 2 < KV) ? ap[2] : 0.0f;
          e3 = (gk + 3 < KV) ? ap[3] : 0.0f;
        }
        uint2 p;
        p.x = pack_bf16(e0, e1);
        p.y = pack_bf16(e2, e3);
        *(uint2*)&AsU[r][ck >> 1] = p;              // ds_store_b64
      }

      // ---- stage B tile (32 K x 128 N) transposed into [col][kpair] ----
      for (int t = tid; t < 512; t += 256) {
        const int kp = t >> 5;          // 0..15  (k pair index)
        const int nc = (t & 31) * 4;    // 0..124 (column chunk)
        const int gk = k0 + kp * 2;
        const float* w0 = Wp + (size_t)gk * ldw + wcol0 + blockN + nc;
        float4 q0, q1;
        if constexpr (KV % 32 == 0) {
          q0 = *(const float4*)w0;                  // global_load_b128
          q1 = *(const float4*)(w0 + ldw);
        } else {
          q0 = (gk + 0 < KV) ? *(const float4*)w0
                             : make_float4(0.f, 0.f, 0.f, 0.f);
          q1 = (gk + 1 < KV) ? *(const float4*)(w0 + ldw)
                             : make_float4(0.f, 0.f, 0.f, 0.f);
        }
        BsU[nc + 0][kp] = pack_bf16(q0.x, q1.x);
        BsU[nc + 1][kp] = pack_bf16(q0.y, q1.y);
        BsU[nc + 2][kp] = pack_bf16(q0.z, q1.z);
        BsU[nc + 3][kp] = pack_bf16(q0.w, q1.w);
      }

      if (kt + 1 < kTiles) {            // global_prefetch_b8 for next K tile
        __builtin_prefetch(&Ap[(size_t)(blockM + (tid >> 2)) * KV + k0 + 32], 0, 1);
        __builtin_prefetch(&Wp[(size_t)(k0 + 32 + (tid >> 3)) * ldw + wcol0 + blockN], 0, 1);
      }
      __syncthreads();

      // ---- fragments: 2 x ds_load_b128 each (ISA 7.12.2 wave32 layouts) ----
      const int half = lane >> 4;
      const int ml   = lane & 15;
      v16bf af[2], bf[2];
#pragma unroll
      for (int r = 0; r < 2; ++r) {
        const int row = wm * 32 + r * 16 + ml;
        uint4 lo = *(const uint4*)&AsU[row][half * 4];
        uint4 hi = *(const uint4*)&AsU[row][8 + half * 4];
        FragCast fc; fc.u = (v8u){lo.x, lo.y, lo.z, lo.w, hi.x, hi.y, hi.z, hi.w};
        af[r] = fc.h;
      }
#pragma unroll
      for (int s = 0; s < 2; ++s) {
        const int colr = wn * 32 + s * 16 + ml;
        uint4 lo = *(const uint4*)&BsU[colr][half * 4];
        uint4 hi = *(const uint4*)&BsU[colr][8 + half * 4];
        FragCast fc; fc.u = (v8u){lo.x, lo.y, lo.z, lo.w, hi.x, hi.y, hi.z, hi.w};
        bf[s] = fc.h;
      }

#pragma unroll
      for (int r = 0; r < 2; ++r)
#pragma unroll
        for (int s = 0; s < 2; ++s)
          acc[r][s] = __builtin_amdgcn_wmma_f32_16x16x32_bf16(
              false, af[r], false, bf[s], (short)0, acc[r][s], false, false);
      __syncthreads();
    }
  }

  // ---- epilogue: C layout VGPR t -> M = t + 8*half, lanes&15 -> N ----
  const int half = lane >> 4;
  const int ml   = lane & 15;
#pragma unroll
  for (int s = 0; s < 2; ++s) {
    const int col = blockN + wn * 32 + s * 16 + ml;
    float bv = 0.0f;
    if constexpr (HASBIAS) bv = bias[col];
#pragma unroll
    for (int r = 0; r < 2; ++r) {
      const int rbase = blockM + wm * 32 + r * 16 + half * 8;
#pragma unroll
      for (int t = 0; t < 8; ++t) {
        size_t idx = (size_t)(rbase + t) * N + col;
        float v = acc[r][s][t] + bv;
        if constexpr (GELU) v = gelu_f(v);
        Cmat[idx] = v;
      }
    }
  }
}

// ---------------------------------------------------------------------------
// segment ops (graph aggregation / pooling)
// ---------------------------------------------------------------------------
__global__ void seg_count(const int* __restrict__ seg, float* __restrict__ cnt, int n) {
  int i = blockIdx.x * blockDim.x + threadIdx.x;
  if (i < n) atomicAdd(&cnt[seg[i]], 1.0f);
}

__global__ void seg_scatter_add(const float* __restrict__ X, const int* __restrict__ src,
                                const int* __restrict__ dst, float* __restrict__ out,
                                int n, int D) {
  long long i = (long long)blockIdx.x * blockDim.x + threadIdx.x;
  if (i >= (long long)n * D) return;
  int e = (int)(i / D), d = (int)(i % D);
  int s = src ? src[e] : e;
  atomicAdd(&out[(long long)dst[e] * D + d], X[(long long)s * D + d]);
}

__global__ void seg_normalize(float* __restrict__ X, const float* __restrict__ cnt,
                              int rows, int D) {
  long long i = (long long)blockIdx.x * blockDim.x + threadIdx.x;
  if (i >= (long long)rows * D) return;
  int r = (int)(i / D);
  X[i] = X[i] / fmaxf(cnt[r], 1.0f);
}

// ---------------------------------------------------------------------------
// per-row layernorm (+optional exact GELU); blockDim.x == cols (pow2 <=256)
// ---------------------------------------------------------------------------
__global__ void ln_rows(const float* __restrict__ X, const float* __restrict__ g,
                        const float* __restrict__ be, float* __restrict__ Y,
                        int cols, int apply_gelu) {
  __shared__ float red[256];
  const int row = blockIdx.x, tid = threadIdx.x;
  float x = X[(size_t)row * cols + tid];
  red[tid] = x; __syncthreads();
  for (int s = cols >> 1; s > 0; s >>= 1) {
    if (tid < s) red[tid] += red[tid + s];
    __syncthreads();
  }
  float mean = red[0] / (float)cols; __syncthreads();
  float d = x - mean;
  red[tid] = d * d; __syncthreads();
  for (int s = cols >> 1; s > 0; s >>= 1) {
    if (tid < s) red[tid] += red[tid + s];
    __syncthreads();
  }
  float var = red[0] / (float)cols;
  float y = d * rsqrtf(var + LN_EPS) * g[tid] + be[tid];
  if (apply_gelu) y = gelu_f(y);
  Y[(size_t)row * cols + tid] = y;
}

// ---------------------------------------------------------------------------
// VQ: code norms, argmin, gather
// ---------------------------------------------------------------------------
__global__ void qc_norms(const float* __restrict__ qc, float* __restrict__ qcn, int k) {
  int i = blockIdx.x * blockDim.x + threadIdx.x;
  if (i >= k) return;
  const float* q = qc + (size_t)i * EMB;
  float s = 0.f;
  for (int c = 0; c < EMB; ++c) s += q[c] * q[c];
  qcn[i] = s;
}

__global__ __launch_bounds__(256) void vq_argmin(
    const float* __restrict__ z, const float* __restrict__ qc,
    const float* __restrict__ qcn, int* __restrict__ indices,
    float* __restrict__ quant) {
  const int b = blockIdx.x, tid = threadIdx.x;
  __shared__ float zsh[EMB];
  if (tid < EMB) zsh[tid] = z[b * EMB + tid];
  __syncthreads();
  float best = 1e30f; int bi = 0;
  for (int k = tid; k < KCODE; k += 256) {
    const float* q = qc + (size_t)k * EMB;
    float dot = 0.f;
    for (int c = 0; c < EMB; ++c) dot = fmaf(zsh[c], q[c], dot);
    float d2 = qcn[k] - 2.0f * dot;   // + const ||z||^2 irrelevant for argmin
    if (d2 < best) { best = d2; bi = k; }
  }
  __shared__ float bv[256]; __shared__ int bx[256];
  bv[tid] = best; bx[tid] = bi; __syncthreads();
  for (int s = 128; s > 0; s >>= 1) {
    if (tid < s) {
      if (bv[tid + s] < bv[tid] ||
          (bv[tid + s] == bv[tid] && bx[tid + s] < bx[tid])) {
        bv[tid] = bv[tid + s]; bx[tid] = bx[tid + s];
      }
    }
    __syncthreads();
  }
  int bestk = bx[0];
  if (tid == 0) indices[b] = bestk;
  if (tid < EMB) quant[b * EMB + tid] = qc[(size_t)bestk * EMB + tid];
}

// ---------------------------------------------------------------------------
// decoder helpers
// ---------------------------------------------------------------------------
__global__ void build_hd(const float* __restrict__ v, const float* __restrict__ q,
                         float* __restrict__ hd) {
  int r = blockIdx.x, c = threadIdx.x;
  int b = r >> 7, m = r & 127;
  hd[(size_t)r * EMB + c] = v[b * EMB + c] + q[m * EMB + c];
}

__global__ void coords_mask(const float* __restrict__ hm, const float* __restrict__ W3,
                            const float* __restrict__ b3, const int* __restrict__ nvert,
                            float* __restrict__ out) {
  int r = blockIdx.x * blockDim.x + threadIdx.x;
  if (r >= BB * MAXV) return;
  int b = r >> 7, m = r & 127;
  bool on = m < nvert[b];
  const float* row = hm + (size_t)r * 256;
  for (int d = 0; d < 3; ++d) {
    float s = b3[d];
    for (int c = 0; c < 256; ++c) s = fmaf(row[c], W3[c * 3 + d], s);
    out[r * 3 + d] = on ? s : 0.0f;
  }
}

// ---------------------------------------------------------------------------
// losses
// ---------------------------------------------------------------------------
__global__ __launch_bounds__(128) void chamfer(const float* __restrict__ recon,
                                               const float* __restrict__ gt,
                                               const int* __restrict__ nvert,
                                               float* __restrict__ acc) {
  const int b = blockIdx.x, tid = threadIdx.x;
  __shared__ float sr[MAXV][3], sg[MAXV][3], red[128];
  int nv = nvert[b];
  for (int d = 0; d < 3; ++d) {
    sr[tid][d] = recon[((size_t)b * MAXV + tid) * 3 + d];
    sg[tid][d] = gt[((size_t)b * MAXV + tid) * 3 + d];
  }
  __syncthreads();
  float mnr = 0.f, mng = 0.f;
  if (tid < nv) {
    float m1 = 1e30f, m2 = 1e30f;
    for (int j = 0; j < nv; ++j) {
      float dx = sr[tid][0] - sg[j][0];
      float dy = sr[tid][1] - sg[j][1];
      float dz = sr[tid][2] - sg[j][2];
      m1 = fminf(m1, dx * dx + dy * dy + dz * dz);
      float ex = sr[j][0] - sg[tid][0];
      float ey = sr[j][1] - sg[tid][1];
      float ez = sr[j][2] - sg[tid][2];
      m2 = fminf(m2, ex * ex + ey * ey + ez * ez);
    }
    mnr = m1; mng = m2;
  }
  red[tid] = mnr + mng; __syncthreads();
  for (int s = 64; s > 0; s >>= 1) {
    if (tid < s) red[tid] += red[tid + s];
    __syncthreads();
  }
  if (tid == 0)
    atomicAdd(acc, red[0] / (float)(nv > 1 ? nv : 1) * (1.0f / (float)BB));
}

__global__ __launch_bounds__(256) void vq_loss(const float* __restrict__ z,
                                               const float* __restrict__ q,
                                               float* __restrict__ acc) {
  int i = blockIdx.x * 256 + threadIdx.x;   // grid covers BB*EMB = 32768
  float d = z[i] - q[i];
  __shared__ float red[256];
  red[threadIdx.x] = d * d; __syncthreads();
  for (int s = 128; s > 0; s >>= 1) {
    if (threadIdx.x < s) red[threadIdx.x] += red[threadIdx.x + s];
    __syncthreads();
  }
  if (threadIdx.x == 0) atomicAdd(acc, red[0] * (1.0f / (float)(BB * EMB)));
}

__global__ void finalize(const float* __restrict__ accs, const int* __restrict__ idx,
                         float* __restrict__ out) {
  int t = threadIdx.x;
  if (t == 0) {
    float rl = accs[0], cl = accs[1];
    out[RECON_N + 0] = rl + 2.0f * cl;  // total
    out[RECON_N + 1] = rl;              // recon
    out[RECON_N + 2] = cl;              // commit
    out[RECON_N + 3] = cl;              // embed (identical in fwd)
  }
  if (t < BB) out[RECON_N + 4 + t] = (float)idx[t];
}

// ---------------------------------------------------------------------------
extern "C" void kernel_launch(void* const* d_in, const int* in_sizes, int n_in,
                              void* d_out, int out_size, void* d_ws, size_t ws_size,
                              hipStream_t stream) {
  const float *x, *gt, *Wl[4], *bl[4], *Wr[4], *gG[4], *gB[4];
  const float *W_vq, *queries, *in_proj_W, *in_proj_b, *out_W, *out_b;
  const float *dec_g, *dec_beta, *mlp_W1, *mlp_b1, *mlp_W2, *mlp_b2, *mlp_W3, *mlp_b3, *Ccode;
  const int *edge, *batch, *nvert;

  static const int di_[4] = {15, 256, 256, 256};
  static const int do_[4] = {256, 256, 256, 128};

  if (n_in >= 40) {
    x = (const float*)d_in[0]; gt = (const float*)d_in[1];
    for (int l = 0; l < 4; ++l) {
      Wl[l] = (const float*)d_in[2 + l];
      bl[l] = (const float*)d_in[6 + l];
      Wr[l] = (const float*)d_in[10 + l];
      gG[l] = (const float*)d_in[14 + l];
      gB[l] = (const float*)d_in[18 + l];
    }
    W_vq = (const float*)d_in[22]; queries = (const float*)d_in[23];
    in_proj_W = (const float*)d_in[24]; in_proj_b = (const float*)d_in[25];
    out_W = (const float*)d_in[26]; out_b = (const float*)d_in[27];
    dec_g = (const float*)d_in[28]; dec_beta = (const float*)d_in[29];
    mlp_W1 = (const float*)d_in[30]; mlp_b1 = (const float*)d_in[31];
    mlp_W2 = (const float*)d_in[32]; mlp_b2 = (const float*)d_in[33];
    mlp_W3 = (const float*)d_in[34]; mlp_b3 = (const float*)d_in[35];
    Ccode = (const float*)d_in[36];
    edge = (const int*)d_in[37]; batch = (const int*)d_in[38]; nvert = (const int*)d_in[39];
  } else {
    x = (const float*)d_in[0]; gt = (const float*)d_in[1];
    const float* wlc = (const float*)d_in[2];
    const float* blc = (const float*)d_in[3];
    const float* wrc = (const float*)d_in[4];
    const float* gc  = (const float*)d_in[5];
    const float* bc  = (const float*)d_in[6];
    size_t wo = 0, bo = 0;
    for (int l = 0; l < 4; ++l) {
      Wl[l] = wlc + wo; Wr[l] = wrc + wo;
      bl[l] = blc + bo; gG[l] = gc + bo; gB[l] = bc + bo;
      wo += (size_t)di_[l] * do_[l]; bo += do_[l];
    }
    W_vq = (const float*)d_in[7]; queries = (const float*)d_in[8];
    in_proj_W = (const float*)d_in[9]; in_proj_b = (const float*)d_in[10];
    out_W = (const float*)d_in[11]; out_b = (const float*)d_in[12];
    dec_g = (const float*)d_in[13]; dec_beta = (const float*)d_in[14];
    mlp_W1 = (const float*)d_in[15]; mlp_b1 = (const float*)d_in[16];
    mlp_W2 = (const float*)d_in[17]; mlp_b2 = (const float*)d_in[18];
    mlp_W3 = (const float*)d_in[19]; mlp_b3 = (const float*)d_in[20];
    Ccode = (const float*)d_in[21];
    edge = (const int*)d_in[22]; batch = (const int*)d_in[23]; nvert = (const int*)d_in[24];
  }
  const int* srcp = edge;
  const int* dstp = edge + N_EDGES;
  float* dof = (float*)d_out;

  // ---- workspace carve-up
  char* ws = (char*)d_ws;
  size_t off = 0;
  auto carve = [&](size_t bytes) -> char* {
    char* p = ws + off;
    off = (off + bytes + 255) & ~(size_t)255;
    return p;
  };
  const size_t nodeBuf = (size_t)N_NODES * 256 * sizeof(float);
  float* bufA  = (float*)carve(nodeBuf);
  float* bufB  = (float*)carve(nodeBuf);
  float* agg   = (float*)carve(nodeBuf);
  float* cnt   = (float*)carve((size_t)N_NODES * sizeof(float));
  float* z     = (float*)carve((size_t)BB * EMB * sizeof(float));
  float* cntb  = (float*)carve((size_t)BB * sizeof(float));
  float* qc    = (float*)carve((size_t)KCODE * EMB * sizeof(float));
  float* qcn   = (float*)carve((size_t)KCODE * sizeof(float));
  float* quant = (float*)carve((size_t)BB * EMB * sizeof(float));
  int*   idx   = (int*)  carve((size_t)BB * sizeof(int));
  float* t1    = (float*)carve((size_t)BB * EMB * sizeof(float));
  float* vbuf  = (float*)carve((size_t)BB * EMB * sizeof(float));
  float* accs  = (float*)carve(256);
  // decoder tensors reuse freed node buffers (final h lives in bufB)
  float* hd  = bufA;                                   // 32768*128 f32
  float* hm1 = agg;                                    // 32768*256 f32
  float* hm2 = (float*)((char*)bufA + 33554432);       // disjoint from hd

  // ================= encoder: 4 fused SAGE layers =================
  // out = agg @ Wl + h @ Wr + bl  (single dual-A WMMA GEMM), then LN+GELU
  const float* cur = x;
  float* outs[4] = {bufA, bufB, bufA, bufB};
  for (int l = 0; l < 4; ++l) {
    const int di = di_[l], dn = do_[l];
    hipMemsetAsync(agg, 0, (size_t)N_NODES * di * sizeof(float), stream);
    hipMemsetAsync(cnt, 0, (size_t)N_NODES * sizeof(float), stream);
    seg_count<<<(N_EDGES + 255) / 256, 256, 0, stream>>>(dstp, cnt, N_EDGES);
    {
      long long tot = (long long)N_EDGES * di;
      seg_scatter_add<<<(unsigned)((tot + 255) / 256), 256, 0, stream>>>(
          cur, srcp, dstp, agg, N_EDGES, di);
    }
    {
      long long tot = (long long)N_NODES * di;
      seg_normalize<<<(unsigned)((tot + 255) / 256), 256, 0, stream>>>(agg, cnt, N_NODES, di);
    }
    float* outp = outs[l];
    dim3 grid(dn / 128, N_NODES / 64);
    if (di == 15)
      gemm_bf16_wmma<15, true, true, false><<<grid, 256, 0, stream>>>(
          agg, Wl[l], cur, Wr[l], bl[l], outp, N_NODES, dn, dn, 0);
    else
      gemm_bf16_wmma<256, true, true, false><<<grid, 256, 0, stream>>>(
          agg, Wl[l], cur, Wr[l], bl[l], outp, N_NODES, dn, dn, 0);
    ln_rows<<<N_NODES, dn, 0, stream>>>(outp, gG[l], gB[l], outp, dn, 1);
    cur = outp;
  }
  const float* h_final = cur;   // bufB, N x 128

  // ================= pooling =================
  hipMemsetAsync(z, 0, (size_t)BB * EMB * sizeof(float), stream);
  hipMemsetAsync(cntb, 0, (size_t)BB * sizeof(float), stream);
  seg_count<<<(N_NODES + 255) / 256, 256, 0, stream>>>(batch, cntb, N_NODES);
  {
    long long tot = (long long)N_NODES * EMB;
    seg_scatter_add<<<(unsigned)((tot + 255) / 256), 256, 0, stream>>>(
        h_final, nullptr, batch, z, N_NODES, EMB);
    seg_normalize<<<(unsigned)((BB * EMB + 255) / 256), 256, 0, stream>>>(z, cntb, BB, EMB);
  }

  // ================= vector quantization =================
  {
    dim3 grid(EMB / 128, KCODE / 64);
    gemm_bf16_wmma<128, false, false, false><<<grid, 256, 0, stream>>>(
        Ccode, W_vq, nullptr, nullptr, nullptr, qc, KCODE, EMB, EMB, 0);
  }
  qc_norms<<<(KCODE + 255) / 256, 256, 0, stream>>>(qc, qcn, KCODE);
  vq_argmin<<<BB, 256, 0, stream>>>(z, qc, qcn, idx, quant);

  // v = (quant @ Wv + bv) @ out_W + out_b   (Wv = in_proj_W[:, 2E:3E])
  {
    dim3 grid(EMB / 128, BB / 64);
    gemm_bf16_wmma<128, false, true, false><<<grid, 256, 0, stream>>>(
        quant, in_proj_W, nullptr, nullptr, in_proj_b + 2 * EMB, t1, BB, EMB, 3 * EMB, 2 * EMB);
    gemm_bf16_wmma<128, false, true, false><<<grid, 256, 0, stream>>>(
        t1, out_W, nullptr, nullptr, out_b, vbuf, BB, EMB, EMB, 0);
  }

  // ================= decoder =================
  build_hd<<<BB * MAXV, EMB, 0, stream>>>(vbuf, queries, hd);
  ln_rows<<<BB * MAXV, EMB, 0, stream>>>(hd, dec_g, dec_beta, hd, EMB, 0);
  {
    dim3 grid(256 / 128, (BB * MAXV) / 64);
    gemm_bf16_wmma<128, false, true, true><<<grid, 256, 0, stream>>>(
        hd, mlp_W1, nullptr, nullptr, mlp_b1, hm1, BB * MAXV, 256, 256, 0);
    gemm_bf16_wmma<256, false, true, true><<<grid, 256, 0, stream>>>(
        hm1, mlp_W2, nullptr, nullptr, mlp_b2, hm2, BB * MAXV, 256, 256, 0);
  }
  coords_mask<<<(BB * MAXV + 255) / 256, 256, 0, stream>>>(hm2, mlp_W3, mlp_b3, nvert, dof);

  // ================= losses + outputs =================
  hipMemsetAsync(accs, 0, 256, stream);
  chamfer<<<BB, 128, 0, stream>>>(dof, gt, nvert, &accs[0]);
  vq_loss<<<(BB * EMB) / 256, 256, 0, stream>>>(z, quant, &accs[1]);
  finalize<<<1, 256, 0, stream>>>(accs, idx, dof);
}
